// CountHistogram_49108656062551
// MI455X (gfx1250) — compile-verified, matrix operand order checked by reading, not run on hardware
//
#include <hip/hip_runtime.h>
#include <hip/hip_bf16.h>

typedef __attribute__((ext_vector_type(16))) _Float16 v16h;
typedef __attribute__((ext_vector_type(8)))  float    v8f;

#define NBINS      29
#define NBINS_PAD  32
#define LD         8192
#define THREADS    256
#define COPIES     256   // one private histogram copy per thread

// One workgroup per (b,q) row. Bin-major LDS layout sh[bin*COPIES + tid]:
// DS bank = tid % 64 -> all 32 lanes of a wave hit distinct banks for any bin
// pattern, so ds_add_f32 to the thread-private slot is conflict-free.
__global__ __launch_bounds__(THREADS)
void CountHistogram_kernel(const float* __restrict__ sim,
                           const int*  __restrict__ msk,
                           float* __restrict__ out) {
  __shared__ float sh[NBINS_PAD * COPIES];   // 32 KB
  const int tid = threadIdx.x;
  const int row = blockIdx.x;

  // Zero private histograms: 8192 floats / 256 threads = 32 each.
  #pragma unroll
  for (int i = 0; i < NBINS_PAD; ++i)
    sh[i * COPIES + tid] = 0.0f;
  __syncthreads();

  // ---- Main streaming loop: 8192 elems/row as 2048 float4/int4 vectors ----
  const float4* __restrict__ simv = (const float4*)(sim + (size_t)row * LD);
  const int4*   __restrict__ mskv = (const int4*)(msk + (size_t)row * LD);

  #pragma unroll
  for (int it = 0; it < LD / (4 * THREADS); ++it) {
    const int vi = it * THREADS + tid;
    const float4 s4 = simv[vi];
    const int4   m4 = mskv[vi];

    // bin = (int)((s + 1.00001) / 2 * (NBINS-1)) == (int)(s*14 + 14.00014)
    #define HIST_STEP(SV, MV)                                          \
      if (MV) {                                                        \
        int bin = (int)fmaf((SV), 14.0f, 14.00014f);                   \
        bin = (bin < 0) ? 0 : ((bin > NBINS_PAD - 1) ? NBINS_PAD - 1 : bin); \
        atomicAdd(&sh[bin * COPIES + tid], 1.0f);                      \
      }
    HIST_STEP(s4.x, m4.x)
    HIST_STEP(s4.y, m4.y)
    HIST_STEP(s4.z, m4.z)
    HIST_STEP(s4.w, m4.w)
    #undef HIST_STEP
  }

  // ---- Fold 256 copies -> 32 copies (values stay <= 256, exact in f16) ----
  __syncthreads();
  for (int i = tid; i < NBINS_PAD * 128; i += THREADS) {
    int b = i >> 7, c = i & 127;
    sh[b * COPIES + c] += sh[b * COPIES + c + 128];
  }
  __syncthreads();
  for (int i = tid; i < NBINS_PAD * 64; i += THREADS) {
    int b = i >> 6, c = i & 63;
    sh[b * COPIES + c] += sh[b * COPIES + c + 64];
  }
  __syncthreads();
  for (int i = tid; i < NBINS_PAD * 32; i += THREADS) {
    int b = i >> 5, c = i & 31;
    sh[b * COPIES + c] += sh[b * COPIES + c + 32];
  }
  __syncthreads();

  // ---- Final 32-way reduction as a ones-GEMM on the WMMA unit ----
  // D[m,n] = sum_k 1 * B[k,n], B[k,n] = partial copy k of bin n.
  // Wave 0 only (uniform branch -> EXEC all ones inside).
  if (tid < 32) {
    const int lane = tid;
    const int col  = lane & 15;               // B column = bin (within group)
    const int kb   = (lane < 16) ? 0 : 8;     // K rows held by this lane half

    v16h a, b0, b1;
    #pragma unroll
    for (int i = 0; i < 8; ++i) {
      a[i]     = (_Float16)1.0f;
      a[i + 8] = (_Float16)1.0f;
      // lane<16 supplies K = {0..7, 16..23}; lane>=16 supplies {8..15, 24..31}
      b0[i]     = (_Float16)sh[col * COPIES + (kb + i)];
      b0[i + 8] = (_Float16)sh[col * COPIES + (kb + 16 + i)];
      b1[i]     = (_Float16)sh[(col + 16) * COPIES + (kb + i)];
      b1[i + 8] = (_Float16)sh[(col + 16) * COPIES + (kb + 16 + i)];
    }

    v8f d0 = {};
    v8f d1 = {};
    d0 = __builtin_amdgcn_wmma_f32_16x16x32_f16(false, a, false, b0,
                                                (short)0, d0, false, false);
    d1 = __builtin_amdgcn_wmma_f32_16x16x32_f16(false, a, false, b1,
                                                (short)0, d1, false, false);

    // D VGPR0, lanes 0..15 hold (M=0, N=lane) -> the 16 column sums.
    if (lane < 16) {
      float* orow = out + (size_t)row * NBINS;
      orow[lane] = d0[0];                 // bins 0..15
      if (lane < NBINS - 16)              // bins 16..28
        orow[16 + lane] = d1[0];
    }
  }
}

extern "C" void kernel_launch(void* const* d_in, const int* in_sizes, int n_in,
                              void* d_out, int out_size, void* d_ws, size_t ws_size,
                              hipStream_t stream) {
  const float* sim = (const float*)d_in[0];   // simmat (B, Lq, Ld) f32
  // d_in[1] = dlens (unused by the math)
  const int*   msk = (const int*)d_in[2];     // mask   (B, Lq, Ld) int
  float* out = (float*)d_out;                 // (B, Lq, NBINS) f32

  const int rows = in_sizes[0] / LD;          // B * Lq = 2048
  CountHistogram_kernel<<<rows, THREADS, 0, stream>>>(sim, msk, out);
}
